// SCAM_71038759075973
// MI455X (gfx1250) — matrix operand order repaired and checked by reading, not compiled, not run on hardware
//
#include <hip/hip_runtime.h>

// ---------------------------------------------------------------------------
// SCAM (stereo cross-attention) fused for MI455X / gfx1250, wave32 + WMMA bf16.
// One workgroup per (b, h): all eight 128x128 GEMMs of that row pair stay in
// LDS (304KB of the 320KB WGP budget). fp32 accumulate, bf16 operands.
// Weight staging into LDS uses the CDNA5 async global->LDS copy engine
// (GLOBAL_LOAD_ASYNC_TO_LDS_B128, ASYNCcnt) so no VGPR round-trip competes
// with the WMMA pipeline.
// ---------------------------------------------------------------------------

typedef __bf16 bf16;
typedef __attribute__((ext_vector_type(8)))  bf16  v8bf;
typedef __attribute__((ext_vector_type(16))) bf16  v16bf;
typedef __attribute__((ext_vector_type(8)))  float v8f;

#define PB 136   // bf16 LDS row pitch (elements): 272B = 16*17, 16B aligned
#define PF 132   // fp32 LDS row pitch (elements)

// Async 16B copy global -> LDS (per enabled lane). GV mode, saddr = off.
// lds_off is a byte offset from the workgroup's dynamic-LDS base (offset 0:
// these kernels have no static __shared__).
__device__ __forceinline__ void async_copy_b128(unsigned lds_off, const void* gaddr) {
  asm volatile("global_load_async_to_lds_b128 %0, %1, off"
               :
               : "v"(lds_off), "v"(gaddr)
               : "memory");
}
__device__ __forceinline__ void async_wait0() {
  asm volatile("s_wait_asynccnt 0x0" ::: "memory");
}

// A-operand fragment, 16x32 bf16, ISA layout:
//   lane L: row m = mt*16 + (L&15); K runs [koff..koff+7], [koff+16..koff+23],
//   koff = (L>>4)*8.  Two 16B loads.
__device__ __forceinline__ v16bf frag_a(const bf16* base, int pitch, int mt, int kt, int lane) {
  const bf16* p = base + (mt * 16 + (lane & 15)) * pitch + kt * 32 + ((lane >> 4) << 3);
  v8bf lo = *(const v8bf*)(p);
  v8bf hi = *(const v8bf*)(p + 16);
  v16bf r;
#pragma unroll
  for (int i = 0; i < 8; ++i) { r[i] = lo[i]; r[i + 8] = hi[i]; }
  return r;
}

// B-operand fragment, 32x16 bf16, stored K-major as Bt[n][k]:
//   lane L: col n = nt*16 + (L&15); K runs [koff..koff+7], [koff+16..koff+23],
//   koff = (L>>4)*8. Two 16B loads.
__device__ __forceinline__ v16bf frag_b(const bf16* base, int pitch, int nt, int kt, int lane) {
  const bf16* p = base + (nt * 16 + (lane & 15)) * pitch + kt * 32 + ((lane >> 4) << 3);
  v8bf lo = *(const v8bf*)(p);
  v8bf hi = *(const v8bf*)(p + 16);
  v16bf r;
#pragma unroll
  for (int i = 0; i < 8; ++i) { r[i] = lo[i]; r[i + 8] = hi[i]; }
  return r;
}

// 128xN GEMM slice: this wave owns M-block `wave`, loops 8 N-tiles, KT K-steps.
template <int KT, typename EPI>
__device__ __forceinline__ void gemm8x8(const bf16* A, int pa, const bf16* Bt, int pb,
                                        int lane, int wave, EPI epi) {
#pragma unroll
  for (int nt = 0; nt < 8; ++nt) {
    v8f acc = {};
#pragma unroll
    for (int kt = 0; kt < KT; ++kt) {
      v16bf a = frag_a(A, pa, wave, kt, lane);
      v16bf b = frag_b(Bt, pb, nt, kt, lane);
      acc = __builtin_amdgcn_wmma_f32_16x16x32_bf16(false, a, false, b,
                                                    (short)0, acc, false, false);
    }
    epi(nt, acc);
  }
}

// ---- weight fp32 -> bf16 conversion into workspace ------------------------
// layout: [Wm 8192][Wl1 16384][Wr1 16384][Wl2 16384][Wr2 16384][Wf 32768]
__global__ __launch_bounds__(256) void scam_convert(const float* __restrict__ Wm,
                                                    const float* __restrict__ Wl1,
                                                    const float* __restrict__ Wr1,
                                                    const float* __restrict__ Wl2,
                                                    const float* __restrict__ Wr2,
                                                    const float* __restrict__ Wf,
                                                    bf16* __restrict__ dst) {
  int i = blockIdx.x * 256 + threadIdx.x;
  if (i < 8192)            dst[i] = (bf16)Wm[i];
  else if (i < 24576)      dst[i] = (bf16)Wl1[i - 8192];
  else if (i < 40960)      dst[i] = (bf16)Wr1[i - 24576];
  else if (i < 57344)      dst[i] = (bf16)Wl2[i - 40960];
  else if (i < 73728)      dst[i] = (bf16)Wr2[i - 57344];
  else if (i < 106496)     dst[i] = (bf16)Wf[i - 73728];
}

// ---- x_r prologue: bilinear(align_corners) upsample 64x64 -> 128x128, then
// conv1x1 (64->128ch) as a 128x128x64 WMMA GEMM. Interp and conv commute. ---
__global__ __launch_bounds__(256) void scam_prep(const float* __restrict__ xr,
                                                 const bf16* __restrict__ WmB,
                                                 const float* __restrict__ bm,
                                                 float* __restrict__ xr_up) {
  extern __shared__ char smem[];
  bf16* sR  = (bf16*)smem;        // [x=128][mc pitch 72]
  bf16* sWm = sR + 128 * 72;      // staged Wm [o=128][64]
  const unsigned OFF_WM = 128u * 72u * 2u;  // byte offset of sWm in LDS
  const int tid = threadIdx.x, lane = tid & 31, wave = tid >> 5;
  const int b = blockIdx.x >> 7, y = blockIdx.x & 127;
  const float s = 63.0f / 127.0f;
  float cy = (float)y * s;
  int y0 = (int)cy; if (y0 > 62) y0 = 62;
  float fy = cy - (float)y0;

  // stage Wm once per WG via async DMA (16KB = 1024 x 16B)
  for (int i = tid; i < 1024; i += 256)
    async_copy_b128(OFF_WM + i * 16, (const char*)WmB + i * 16);

  for (int idx = tid; idx < 128 * 64; idx += 256) {
    int x = idx >> 6, mc = idx & 63;
    float cx = (float)x * s;
    int x0 = (int)cx; if (x0 > 62) x0 = 62;
    float fx = cx - (float)x0;
    const float* p = xr + ((size_t)(b * 64 + mc) * 64 + y0) * 64 + x0;
    float r00 = p[0], r01 = p[1], r10 = p[64], r11 = p[65];
    float v = (1.0f - fy) * ((1.0f - fx) * r00 + fx * r01) +
              fy * ((1.0f - fx) * r10 + fx * r11);
    sR[x * 72 + mc] = (bf16)v;
  }
  async_wait0();
  __syncthreads();

  // M = out-channel o, N = x, K = mc.  A = sWm[o][64], Bt = sR[x][mc].
  gemm8x8<2>(sWm, 64, sR, 72, lane, wave, [&](int nt, v8f acc) {
    int n = nt * 16 + (lane & 15);
    int m0 = wave * 16 + ((lane >> 4) << 3);
#pragma unroll
    for (int r = 0; r < 8; ++r) {
      int o = m0 + r;
      xr_up[(size_t)b * 2097152 + (size_t)o * 16384 + y * 128 + n] = acc[r] + bm[o];
    }
  });
}

// ---- main fused attention kernel: one WG per (b, h) -----------------------
__global__ __launch_bounds__(256) void scam_main(
    const float* __restrict__ xl_g, const float* __restrict__ xrup_g,
    const bf16* __restrict__ wB, const float* __restrict__ nlw,
    const float* __restrict__ nlb, const float* __restrict__ nrw,
    const float* __restrict__ nrb, const float* __restrict__ bl1,
    const float* __restrict__ br1, const float* __restrict__ bl2,
    const float* __restrict__ br2, const float* __restrict__ beta,
    const float* __restrict__ gamma, const float* __restrict__ bf_,
    float* __restrict__ out) {
  const bf16* Wl1B = wB + 8192;
  const bf16* Wr1B = Wl1B + 16384;
  const bf16* Wl2B = Wr1B + 16384;
  const bf16* Wr2B = Wl2B + 16384;
  const bf16* WfB  = Wr2B + 16384;  // [o][256]

  extern __shared__ char smem[];
  bf16* sXl  = (bf16*)smem;        // [w][c]  x_l row; later Wf staging (w/ sXr)
  bf16* sXr  = sXl + 128 * PB;     // [v][c]  x_r upsampled row
  bf16* sXn  = sXr + 128 * PB;     // LN output [.][c]; later Yl [w][c]
  bf16* sQl  = sXn + 128 * PB;     // Q_l [w][o] -> P1 [w][v] -> Yr [v][c]
  bf16* sQr  = sQl + 128 * PB;     // Q_r [v][o] -> P2T [v][w]
  bf16* sVlT = sQr + 128 * PB;     // V_l transposed [c][w]
  bf16* sVrT = sVlT + 128 * PB;    // V_r transposed [c][v]
  float* sS  = (float*)(sVrT + 128 * PB);  // weight stage -> scores -> out
  bf16*  sW  = (bf16*)sS;          // two staged 128x128 bf16 weights
  bf16*  sWf = sXl;                // staged Wf [o][256] (64KB over sXl+sXr)
  const unsigned OFF_SS = 7u * 128u * PB * 2u;  // byte offset of sS/sW in LDS

  const int tid = threadIdx.x, lane = tid & 31, wave = tid >> 5;
  const int b = blockIdx.x >> 7, h = blockIdx.x & 127;
  const float scale = 0.0883883476483184f;  // 1/sqrt(128)
  const size_t gbase = (size_t)b * 2097152 + (size_t)h * 128;

  // Stage Wl1+Wr1 into the (currently dead) score buffer via async DMA
  // (2 x 32KB = 2 x 2048 x 16B), overlapped with the x_l/x_r load+convert.
  for (int i = tid; i < 2048; i += 256) {
    async_copy_b128(OFF_SS + i * 16, (const char*)Wl1B + i * 16);
    async_copy_b128(OFF_SS + 32768 + i * 16, (const char*)Wr1B + i * 16);
  }
  // Load x_l / x_r rows (coalesced in w), transpose into [pix][c] in LDS.
  for (int idx = tid; idx < 16384; idx += 256) {
    int c = idx >> 7, w = idx & 127;
    sXl[w * PB + c] = (bf16)xl_g[gbase + (size_t)c * 16384 + w];
    sXr[w * PB + c] = (bf16)xrup_g[gbase + (size_t)c * 16384 + w];
  }
  async_wait0();
  __syncthreads();

  // LayerNorm over channels (left), one thread per pixel w.
  if (tid < 128) {
    int w = tid;
    float s = 0.f, s2 = 0.f;
    for (int c = 0; c < 128; ++c) { float v = (float)sXl[w * PB + c]; s += v; s2 += v * v; }
    float mu = s * 0.0078125f;
    float rs = __frsqrt_rn(s2 * 0.0078125f - mu * mu + 1e-6f);
    for (int c = 0; c < 128; ++c) {
      float v = (float)sXl[w * PB + c];
      sXn[w * PB + c] = (bf16)(((v - mu) * rs) * nlw[c] + nlb[c]);
    }
  }
  __syncthreads();

  // Q_l = LN(x_l) @ Wl1^T + bl1  -> sQl [w][o]
  gemm8x8<4>(sXn, PB, sW, 128, lane, wave, [&](int nt, v8f acc) {
    int n = nt * 16 + (lane & 15), m0 = wave * 16 + ((lane >> 4) << 3);
#pragma unroll
    for (int r = 0; r < 8; ++r) sQl[(m0 + r) * PB + n] = (bf16)(acc[r] + bl1[n]);
  });
  __syncthreads();

  // LayerNorm (right) into sXn.
  if (tid < 128) {
    int w = tid;
    float s = 0.f, s2 = 0.f;
    for (int c = 0; c < 128; ++c) { float v = (float)sXr[w * PB + c]; s += v; s2 += v * v; }
    float mu = s * 0.0078125f;
    float rs = __frsqrt_rn(s2 * 0.0078125f - mu * mu + 1e-6f);
    for (int c = 0; c < 128; ++c) {
      float v = (float)sXr[w * PB + c];
      sXn[w * PB + c] = (bf16)(((v - mu) * rs) * nrw[c] + nrb[c]);
    }
  }
  __syncthreads();

  // Q_r -> sQr [v][o]
  gemm8x8<4>(sXn, PB, sW + 16384, 128, lane, wave, [&](int nt, v8f acc) {
    int n = nt * 16 + (lane & 15), m0 = wave * 16 + ((lane >> 4) << 3);
#pragma unroll
    for (int r = 0; r < 8; ++r) sQr[(m0 + r) * PB + n] = (bf16)(acc[r] + br1[n]);
  });
  __syncthreads();

  // Re-stage: Wl2+Wr2 over the Q-weights (Q GEMMs done), async DMA.
  for (int i = tid; i < 2048; i += 256) {
    async_copy_b128(OFF_SS + i * 16, (const char*)Wl2B + i * 16);
    async_copy_b128(OFF_SS + 32768 + i * 16, (const char*)Wr2B + i * 16);
  }
  async_wait0();
  __syncthreads();

  // V_l -> sVlT [c][w]  (transposed store, K-major for later F_l2r GEMM)
  gemm8x8<4>(sXl, PB, sW, 128, lane, wave, [&](int nt, v8f acc) {
    int n = nt * 16 + (lane & 15), m0 = wave * 16 + ((lane >> 4) << 3);
#pragma unroll
    for (int r = 0; r < 8; ++r) sVlT[n * PB + (m0 + r)] = (bf16)(acc[r] + bl2[n]);
  });
  // V_r -> sVrT [c][v]
  gemm8x8<4>(sXr, PB, sW + 16384, 128, lane, wave, [&](int nt, v8f acc) {
    int n = nt * 16 + (lane & 15), m0 = wave * 16 + ((lane >> 4) << 3);
#pragma unroll
    for (int r = 0; r < 8; ++r) sVrT[n * PB + (m0 + r)] = (bf16)(acc[r] + br2[n]);
  });
  __syncthreads();

  // S[w][v] = scale * Q_l . Q_r   (Bt = sQr natural [v][o] layout)
  gemm8x8<4>(sQl, PB, sQr, PB, lane, wave, [&](int nt, v8f acc) {
    int n = nt * 16 + (lane & 15), m0 = wave * 16 + ((lane >> 4) << 3);
#pragma unroll
    for (int r = 0; r < 8; ++r) sS[(m0 + r) * PF + n] = acc[r] * scale;
  });
  __syncthreads();

  // Softmax: threads 0..127 do rows (P1 over v), 128..255 do cols (P2T over w).
  if (tid < 128) {
    int w = tid;
    float mx = -3.0e38f;
    for (int v = 0; v < 128; ++v) mx = fmaxf(mx, sS[w * PF + v]);
    float sum = 0.f;
    for (int v = 0; v < 128; ++v) {
      float e = __expf(sS[w * PF + v] - mx);
      sum += e;
      sQl[w * PB + v] = (bf16)e;  // P1 unnormalized
    }
    float inv = 1.0f / sum;
    for (int v = 0; v < 128; ++v)
      sQl[w * PB + v] = (bf16)((float)sQl[w * PB + v] * inv);
  } else {
    int v = tid - 128;
    float mx = -3.0e38f;
    for (int w = 0; w < 128; ++w) mx = fmaxf(mx, sS[w * PF + v]);
    float sum = 0.f;
    for (int w = 0; w < 128; ++w) {
      float e = __expf(sS[w * PF + v] - mx);
      sum += e;
      sQr[v * PB + w] = (bf16)e;  // P2 transposed, unnormalized
    }
    float inv = 1.0f / sum;
    for (int w = 0; w < 128; ++w)
      sQr[v * PB + w] = (bf16)((float)sQr[v * PB + w] * inv);
  }
  __syncthreads();

  // F_r2l = P1 @ V_r; Yl = x_l + beta*F  -> sXn [w][c]
  gemm8x8<4>(sQl, PB, sVrT, PB, lane, wave, [&](int nt, v8f acc) {
    int n = nt * 16 + (lane & 15), m0 = wave * 16 + ((lane >> 4) << 3);
#pragma unroll
    for (int r = 0; r < 8; ++r) {
      int m = m0 + r;
      float y = (float)sXl[m * PB + n] + beta[n] * acc[r];
      sXn[m * PB + n] = (bf16)y;
    }
  });
  __syncthreads();

  // F_l2r = P2^T @ V_l; Yr = x_r + gamma*F -> sQl (P1 dead) [v][c]
  gemm8x8<4>(sQr, PB, sVlT, PB, lane, wave, [&](int nt, v8f acc) {
    int n = nt * 16 + (lane & 15), m0 = wave * 16 + ((lane >> 4) << 3);
#pragma unroll
    for (int r = 0; r < 8; ++r) {
      int m = m0 + r;
      float y = (float)sXr[m * PB + n] + gamma[n] * acc[r];
      sQl[m * PB + n] = (bf16)y;
    }
  });
  __syncthreads();

  // Residuals consumed -> sXl/sXr dead. Stage Wf [o][256] bf16 over them
  // via async DMA (64KB = 4096 x 16B, LDS offset 0).
  for (int i = tid; i < 4096; i += 256)
    async_copy_b128((unsigned)(i * 16), (const char*)WfB + i * 16);
  async_wait0();
  __syncthreads();

  // Final: out[o][w] = Wf[o][0:128].Yl[w] + Wf[o][128:256].Yr[w] + bf
  // K = 256 split across sXn (Yl) and sQl (Yr); Bt = sWf [o][256].
#pragma unroll
  for (int nt = 0; nt < 8; ++nt) {
    v8f acc = {};
#pragma unroll
    for (int kt = 0; kt < 4; ++kt) {
      v16bf a = frag_a(sXn, PB, wave, kt, lane);
      v16bf bfr = frag_b(sWf, 256, nt, kt, lane);
      acc = __builtin_amdgcn_wmma_f32_16x16x32_bf16(false, a, false, bfr,
                                                    (short)0, acc, false, false);
    }
#pragma unroll
    for (int kt = 0; kt < 4; ++kt) {
      v16bf a = frag_a(sQl, PB, wave, kt, lane);
      v16bf bfr = frag_b(sWf, 256, nt, kt + 4, lane);
      acc = __builtin_amdgcn_wmma_f32_16x16x32_bf16(false, a, false, bfr,
                                                    (short)0, acc, false, false);
    }
    int n = nt * 16 + (lane & 15), m0 = wave * 16 + ((lane >> 4) << 3);
#pragma unroll
    for (int r = 0; r < 8; ++r) sS[n * PF + (m0 + r)] = acc[r] + bf_[n];  // [o][w]
  }
  __syncthreads();

  // Coalesced store of out[b, o, h, w].
  for (int idx = tid; idx < 16384; idx += 256) {
    int o = idx >> 7, w = idx & 127;
    out[gbase + (size_t)o * 16384 + w] = sS[o * PF + w];
  }
}

// ---------------------------------------------------------------------------
extern "C" void kernel_launch(void* const* d_in, const int* in_sizes, int n_in,
                              void* d_out, int out_size, void* d_ws, size_t ws_size,
                              hipStream_t stream) {
  const float* x_l  = (const float*)d_in[0];
  const float* x_r  = (const float*)d_in[1];
  const float* Wm   = (const float*)d_in[2];
  const float* bm   = (const float*)d_in[3];
  const float* nlw  = (const float*)d_in[4];
  const float* nlb  = (const float*)d_in[5];
  const float* nrw  = (const float*)d_in[6];
  const float* nrb  = (const float*)d_in[7];
  const float* Wl1  = (const float*)d_in[8];
  const float* bl1  = (const float*)d_in[9];
  const float* Wr1  = (const float*)d_in[10];
  const float* br1  = (const float*)d_in[11];
  const float* Wl2  = (const float*)d_in[12];
  const float* bl2  = (const float*)d_in[13];
  const float* Wr2  = (const float*)d_in[14];
  const float* br2  = (const float*)d_in[15];
  const float* beta = (const float*)d_in[16];
  const float* gam  = (const float*)d_in[17];
  const float* Wf   = (const float*)d_in[18];
  const float* bf_  = (const float*)d_in[19];

  float* xr_up = (float*)d_ws;                                   // 64 MiB fp32
  bf16*  wB    = (bf16*)((char*)d_ws + (size_t)67108864);        // bf16 weights

  scam_convert<<<416, 256, 0, stream>>>(Wm, Wl1, Wr1, Wl2, Wr2, Wf, wB);
  scam_prep<<<1024, 256, (128 * 72 + 128 * 64) * 2, stream>>>(x_r, wB, bm, xr_up);
  scam_main<<<1024, 256, 7 * 128 * PB * 2 + 128 * PF * 4, stream>>>(
      x_l, xr_up, wB, nlw, nlb, nrw, nrb, bl1, br1, bl2, br2, beta, gam, bf_,
      (float*)d_out);
}
